// MultiHeadAttention_43361989820846
// MI455X (gfx1250) — compile-verified, matrix operand order checked by reading, not run on hardware
//
#include <hip/hip_runtime.h>

typedef __attribute__((ext_vector_type(16))) _Float16 v16h;
typedef __attribute__((ext_vector_type(8)))  _Float16 v8h;
typedef __attribute__((ext_vector_type(4)))  _Float16 v4h;
typedef __attribute__((ext_vector_type(8)))  float    v8f;
typedef __attribute__((ext_vector_type(4)))  int      v4i;

#define D_MODEL 1024
#define NUM_HEADS 16
#define D_K 64
#define BATCH 2
#define SEQ 2048
#define MROWS (BATCH * SEQ)  // 4096

#if __has_builtin(__builtin_amdgcn_global_load_async_to_lds_b128)
#define HAVE_ASYNC_LDS 1
#else
#define HAVE_ASYNC_LDS 0
#endif

typedef __attribute__((address_space(1))) v4i* gv4i_ptr;  // global int4
typedef __attribute__((address_space(3))) v4i* lv4i_ptr;  // LDS int4

// ---------------------------------------------------------------------------
// WMMA wrapper: D = A(16x32 f16) * B(32x16 f16) + C(16x16 f32)
// ---------------------------------------------------------------------------
__device__ __forceinline__ v8f wmma_f16(v16h a, v16h b, v8f c) {
  return __builtin_amdgcn_wmma_f32_16x16x32_f16(
      false, a, false, b, (short)0, c, false, false);
}

// Per-lane A/B-operand gather (row = lane&15, K pattern = two contiguous
// 8-half chunks at offsets g*8 and 16+g*8 of a 32-wide K slice, g = lane>>4).
__device__ __forceinline__ v16h load_a_f16(const _Float16* row, int g) {
  v8h lo = *(const v8h*)(row + g * 8);
  v8h hi = *(const v8h*)(row + 16 + g * 8);
  v16h r;
#pragma unroll
  for (int i = 0; i < 8; ++i) { r[i] = lo[i]; r[8 + i] = hi[i]; }
  return r;
}

// Row reductions inside each 16-lane half (wave32: xor<16 stays in-half).
__device__ __forceinline__ float rowmax16(float v) {
#pragma unroll
  for (int m = 1; m < 16; m <<= 1) v = fmaxf(v, __shfl_xor(v, m, 32));
  return v;
}
__device__ __forceinline__ float rowsum16(float v) {
#pragma unroll
  for (int m = 1; m < 16; m <<= 1) v += __shfl_xor(v, m, 32);
  return v;
}

// 16-byte global -> LDS stage: async DMA on gfx1250 (ASYNCcnt-tracked),
// plain copy fallback if the builtin is absent.
__device__ __forceinline__ void stage16(_Float16* lds, const _Float16* g) {
#if HAVE_ASYNC_LDS
  __builtin_amdgcn_global_load_async_to_lds_b128(
      (gv4i_ptr)(g), (lv4i_ptr)(lds), 0, 0);
#else
  *(uint4*)lds = *(const uint4*)g;
#endif
}

__device__ __forceinline__ void wait_stage() {
#if HAVE_ASYNC_LDS
#if __has_builtin(__builtin_amdgcn_s_wait_asynccnt)
  __builtin_amdgcn_s_wait_asynccnt(0);
#else
  asm volatile("s_wait_asynccnt 0" ::: "memory");
#endif
#endif
}

// ---------------------------------------------------------------------------
// Kernel 0: f32 -> f16 conversion (vectorized), done once per tensor so the
// GEMM hot loops carry no cvt chains.
// ---------------------------------------------------------------------------
__global__ __launch_bounds__(256) void cvt_kernel(const float* __restrict__ src,
                                                  _Float16* __restrict__ dst,
                                                  int n4) {
  const int i = blockIdx.x * blockDim.x + threadIdx.x;
  if (i < n4) {
    float4 f = ((const float4*)src)[i];
    v4h h;
    h[0] = (_Float16)f.x; h[1] = (_Float16)f.y;
    h[2] = (_Float16)f.z; h[3] = (_Float16)f.w;
    ((v4h*)dst)[i] = h;
  }
}

// ---------------------------------------------------------------------------
// Kernel 1: Q/K/V projections, out[m,n] = sum_k X[m,k]*W[n,k] + b[n].
// 64x64 output per wave: 16 WMMAs per 32-wide K chunk, 4x reuse of both
// operands.  Q,K stored f16 [b][h][s][d]; V stored transposed [b][h][d][s].
// ---------------------------------------------------------------------------
__global__ __launch_bounds__(256) void proj_kernel(
    const _Float16* __restrict__ xq, const _Float16* __restrict__ xk,
    const _Float16* __restrict__ xv,
    const _Float16* __restrict__ wq, const _Float16* __restrict__ wk,
    const _Float16* __restrict__ wv,
    const float* __restrict__ bq, const float* __restrict__ bk,
    const float* __restrict__ bv,
    _Float16* __restrict__ qh, _Float16* __restrict__ kh,
    _Float16* __restrict__ vt) {
  const int lane = threadIdx.x & 31;
  const int wave = threadIdx.x >> 5;
  const int tile = blockIdx.x * 8 + wave;  // 1024 tiles: 64 (M) x 16 (N)
  const int tn = tile & 15;
  const int tm = tile >> 4;
  const int g = lane >> 4, r = lane & 15;

  const _Float16 *X, *W; const float* bias;
  if (blockIdx.y == 0)      { X = xq; W = wq; bias = bq; }
  else if (blockIdx.y == 1) { X = xk; W = wk; bias = bk; }
  else                      { X = xv; W = wv; bias = bv; }

  v8f c[4][4] = {};
  for (int k0 = 0; k0 < D_MODEL; k0 += 32) {
    v16h a[4];
#pragma unroll
    for (int i = 0; i < 4; ++i)
      a[i] = load_a_f16(X + (size_t)(tm * 64 + i * 16 + r) * D_MODEL + k0, g);
#pragma unroll
    for (int j = 0; j < 4; ++j) {
      v16h b = load_a_f16(W + (size_t)(tn * 64 + j * 16 + r) * D_MODEL + k0, g);
#pragma unroll
      for (int i = 0; i < 4; ++i) c[i][j] = wmma_f16(a[i], b, c[i][j]);
    }
  }

#pragma unroll
  for (int j = 0; j < 4; ++j) {
    const int n = tn * 64 + j * 16 + r;
    const float bn = bias[n];
    const int h = n >> 6, d = n & 63;
#pragma unroll
    for (int i = 0; i < 4; ++i) {
#pragma unroll
      for (int v = 0; v < 8; ++v) {
        const int m = tm * 64 + i * 16 + v + 8 * g;
        const int b_ = m >> 11, s = m & 2047;
        const float val = c[i][j][v] + bn;
        if (blockIdx.y == 2) {
          vt[(((size_t)b_ * NUM_HEADS + h) * D_K + d) * SEQ + s] =
              (_Float16)val;
        } else {
          _Float16* dst = (blockIdx.y == 0) ? qh : kh;
          dst[(((size_t)b_ * NUM_HEADS + h) * SEQ + s) * D_K + d] =
              (_Float16)val;
        }
      }
    }
  }
}

// ---------------------------------------------------------------------------
// Kernel 2: flash attention.  Block = 8 waves = 128 q rows of one (b,h);
// K/V blocks of 32 keys staged once into LDS (async DMA) and shared by all
// 8 waves.  Online softmax, WMMA for QK^T and P*V.
// ---------------------------------------------------------------------------
__global__ __launch_bounds__(256) void attn_kernel(
    const _Float16* __restrict__ qh, const _Float16* __restrict__ kh,
    const _Float16* __restrict__ vt, _Float16* __restrict__ attn_out) {
  __shared__ __align__(16) _Float16 Kbuf[32 * 64];      // 32 keys x 64 d
  __shared__ __align__(16) _Float16 Vbuf[64 * 32];      // 64 d x 32 keys
  __shared__ __align__(16) _Float16 pLds[8][16 * 32];   // per-wave P tile

  const int tid = threadIdx.x;
  const int lane = tid & 31, wave = tid >> 5;
  const int g = lane >> 4, r = lane & 15;
  const int bh = blockIdx.x >> 4;   // 0..31 = b*16+h
  const int qb = blockIdx.x & 15;   // SEQ/128 q-blocks
  const int q0 = qb * 128 + wave * 16;

  const _Float16* qbase = qh + ((size_t)bh * SEQ + q0) * D_K;
  const _Float16* kbase = kh + (size_t)bh * SEQ * D_K;
  const _Float16* vbase = vt + (size_t)bh * D_K * SEQ;

  const v16h aq0 = load_a_f16(qbase + (size_t)r * D_K + 0, g);
  const v16h aq1 = load_a_f16(qbase + (size_t)r * D_K + 32, g);

  v8f o0 = {}, o1 = {}, o2 = {}, o3 = {};
  float mrow[8], lrow[8];
#pragma unroll
  for (int v = 0; v < 8; ++v) { mrow[v] = -INFINITY; lrow[v] = 0.0f; }
  const float scale = 0.125f;  // 1/sqrt(64)

  for (int kb = 0; kb < SEQ; kb += 32) {
    __syncthreads();  // previous block fully consumed before overwrite
    // --- stage K (4KB contiguous) and V^T (64 rows x 32 keys) into LDS ---
    stage16(&Kbuf[tid * 8], kbase + (size_t)kb * D_K + tid * 8);
    {
      const int row = tid >> 2, cch = tid & 3;
      stage16(&Vbuf[row * 32 + cch * 8],
              vbase + (size_t)row * SEQ + kb + cch * 8);
    }
    wait_stage();
    __syncthreads();

    // ---- scores S = Q K^T (two 16x16 tiles over 32 keys) ----
    const v16h bk0a = load_a_f16(&Kbuf[r * 64 + 0], g);
    const v16h bk0b = load_a_f16(&Kbuf[r * 64 + 32], g);
    const v16h bk1a = load_a_f16(&Kbuf[(16 + r) * 64 + 0], g);
    const v16h bk1b = load_a_f16(&Kbuf[(16 + r) * 64 + 32], g);
    v8f s0 = {}, s1 = {};
    s0 = wmma_f16(aq0, bk0a, s0);
    s0 = wmma_f16(aq1, bk0b, s0);
    s1 = wmma_f16(aq0, bk1a, s1);
    s1 = wmma_f16(aq1, bk1b, s1);

    // ---- online softmax; C layout: lane = key col, VGPR v = row v+8g ----
    float alpha[8];
#pragma unroll
    for (int v = 0; v < 8; ++v) {
      const float x0 = s0[v] * scale;
      const float x1 = s1[v] * scale;
      const float bm = rowmax16(fmaxf(x0, x1));
      const float mn = fmaxf(mrow[v], bm);
      const float p0 = __expf(x0 - mn);
      const float p1 = __expf(x1 - mn);
      const float bs = rowsum16(p0 + p1);
      alpha[v] = __expf(mrow[v] - mn);
      lrow[v] = lrow[v] * alpha[v] + bs;
      mrow[v] = mn;
      const int m = v + 8 * g;
      pLds[wave][m * 32 + r]      = (_Float16)p0;
      pLds[wave][m * 32 + 16 + r] = (_Float16)p1;
    }
    // wave-local LDS RAW: hardware keeps per-wave DS ops in order, compiler
    // inserts the dscnt wait for the dependency below.
    const v16h pa = load_a_f16(&pLds[wave][r * 32], g);  // P in A layout

    // ---- rescale running output, then O += P * V ----
#pragma unroll
    for (int v = 0; v < 8; ++v) {
      o0[v] *= alpha[v]; o1[v] *= alpha[v];
      o2[v] *= alpha[v]; o3[v] *= alpha[v];
    }
    const v16h bv0 = load_a_f16(&Vbuf[(0  + r) * 32], g);
    const v16h bv1 = load_a_f16(&Vbuf[(16 + r) * 32], g);
    const v16h bv2 = load_a_f16(&Vbuf[(32 + r) * 32], g);
    const v16h bv3 = load_a_f16(&Vbuf[(48 + r) * 32], g);
    o0 = wmma_f16(pa, bv0, o0);
    o1 = wmma_f16(pa, bv1, o1);
    o2 = wmma_f16(pa, bv2, o2);
    o3 = wmma_f16(pa, bv3, o3);
  }

  // ---- epilogue: normalize and store [b][s][h*64+d] as f16 ----
  const int b = bh >> 4, h = bh & 15;
  _Float16* obase = attn_out + ((size_t)b * SEQ + q0) * D_MODEL + h * D_K;
#pragma unroll
  for (int v = 0; v < 8; ++v) {
    const float inv = 1.0f / lrow[v];
    const size_t rowoff = (size_t)(v + 8 * g) * D_MODEL;
    obase[rowoff + 0 * 16 + r] = (_Float16)(o0[v] * inv);
    obase[rowoff + 1 * 16 + r] = (_Float16)(o1[v] * inv);
    obase[rowoff + 2 * 16 + r] = (_Float16)(o2[v] * inv);
    obase[rowoff + 3 * 16 + r] = (_Float16)(o3[v] * inv);
  }
}

// ---------------------------------------------------------------------------
// Kernel 3: output projection, fp32 result.  Same 64x64 register blocking.
// ---------------------------------------------------------------------------
__global__ __launch_bounds__(256) void outproj_kernel(
    const _Float16* __restrict__ ah, const _Float16* __restrict__ wo,
    const float* __restrict__ bo, float* __restrict__ out) {
  const int lane = threadIdx.x & 31;
  const int wave = threadIdx.x >> 5;
  const int tile = blockIdx.x * 8 + wave;
  const int tn = tile & 15;
  const int tm = tile >> 4;
  const int g = lane >> 4, r = lane & 15;

  v8f c[4][4] = {};
  for (int k0 = 0; k0 < D_MODEL; k0 += 32) {
    v16h a[4];
#pragma unroll
    for (int i = 0; i < 4; ++i)
      a[i] = load_a_f16(ah + (size_t)(tm * 64 + i * 16 + r) * D_MODEL + k0, g);
#pragma unroll
    for (int j = 0; j < 4; ++j) {
      v16h b = load_a_f16(wo + (size_t)(tn * 64 + j * 16 + r) * D_MODEL + k0, g);
#pragma unroll
      for (int i = 0; i < 4; ++i) c[i][j] = wmma_f16(a[i], b, c[i][j]);
    }
  }

#pragma unroll
  for (int j = 0; j < 4; ++j) {
    const int n = tn * 64 + j * 16 + r;
    const float bn = bo[n];
#pragma unroll
    for (int i = 0; i < 4; ++i) {
#pragma unroll
      for (int v = 0; v < 8; ++v) {
        const size_t m = (size_t)(tm * 64 + i * 16 + v + 8 * g);
        out[m * D_MODEL + n] = c[i][j][v] + bn;
      }
    }
  }
}

// ---------------------------------------------------------------------------
extern "C" void kernel_launch(void* const* d_in, const int* in_sizes, int n_in,
                              void* d_out, int out_size, void* d_ws,
                              size_t ws_size, hipStream_t stream) {
  const float* q  = (const float*)d_in[0];
  const float* k  = (const float*)d_in[1];
  const float* v  = (const float*)d_in[2];
  const float* Wq = (const float*)d_in[3];
  const float* bq = (const float*)d_in[4];
  const float* Wk = (const float*)d_in[5];
  const float* bk = (const float*)d_in[6];
  const float* Wv = (const float*)d_in[7];
  const float* bv = (const float*)d_in[8];
  const float* Wo = (const float*)d_in[9];
  const float* bo = (const float*)d_in[10];

  const size_t XEL = (size_t)MROWS * D_MODEL;   // 4M elements
  const size_t WEL = (size_t)D_MODEL * D_MODEL; // 1M elements
  _Float16* p = (_Float16*)d_ws;
  _Float16* xqh = p; p += XEL;
  _Float16* xkh = p; p += XEL;
  _Float16* xvh = p; p += XEL;
  _Float16* wqh = p; p += WEL;
  _Float16* wkh = p; p += WEL;
  _Float16* wvh = p; p += WEL;
  _Float16* woh = p; p += WEL;
  _Float16* qhb = p; p += XEL;
  _Float16* khb = p; p += XEL;
  _Float16* vtb = p; p += XEL;
  _Float16* ahb = p; p += XEL;

  // 0) one-time f32 -> f16 conversions
  cvt_kernel<<<dim3((int)(XEL / 4 / 256)), 256, 0, stream>>>(q, xqh, XEL / 4);
  cvt_kernel<<<dim3((int)(XEL / 4 / 256)), 256, 0, stream>>>(k, xkh, XEL / 4);
  cvt_kernel<<<dim3((int)(XEL / 4 / 256)), 256, 0, stream>>>(v, xvh, XEL / 4);
  cvt_kernel<<<dim3((int)(WEL / 4 / 256)), 256, 0, stream>>>(Wq, wqh, WEL / 4);
  cvt_kernel<<<dim3((int)(WEL / 4 / 256)), 256, 0, stream>>>(Wk, wkh, WEL / 4);
  cvt_kernel<<<dim3((int)(WEL / 4 / 256)), 256, 0, stream>>>(Wv, wvh, WEL / 4);
  cvt_kernel<<<dim3((int)(WEL / 4 / 256)), 256, 0, stream>>>(Wo, woh, WEL / 4);

  // 1) projections: 1024 tiles of 64x64, 8 waves/block, y selects Q/K/V
  proj_kernel<<<dim3(128, 3), 256, 0, stream>>>(xqh, xkh, xvh, wqh, wkh, wvh,
                                                bq, bk, bv, qhb, khb, vtb);
  // 2) flash attention: 512 blocks of 8 waves (128 q rows each)
  attn_kernel<<<dim3(BATCH * NUM_HEADS * (SEQ / 128)), 256, 0, stream>>>(
      qhb, khb, vtb, ahb);
  // 3) output projection
  outproj_kernel<<<dim3(128), 256, 0, stream>>>(ahb, woh, bo, (float*)d_out);
}
